// HUNYUAN_3221225472694
// MI455X (gfx1250) — matrix operand order repaired and checked by reading, not compiled, not run on hardware
//
#include <hip/hip_runtime.h>

// ---------------- model constants ----------------
#define L_    4
#define H_    16
#define KV_   8
#define D_    128
#define DM_   2048
#define FF_   8192
#define V_    32000
#define S_    1024
#define HIST_ 1024
#define T_    2048   // HIST_ + S_
#define EPS_  1e-6f

// ---------------- WMMA types ----------------
typedef __attribute__((ext_vector_type(16))) __bf16 v16bf;
typedef __attribute__((ext_vector_type(8)))  float  v8f;

union FragAB { v16bf v; unsigned int u[8]; };
union FragC  { v8f   v; float f[8]; };

// ---------------- bf16 helpers (manual RNE) ----------------
__device__ __forceinline__ unsigned short f2bf(float f) {
  unsigned u = __float_as_uint(f);
  u += 0x7fffu + ((u >> 16) & 1u);
  return (unsigned short)(u >> 16);
}
__device__ __forceinline__ float bf2f(unsigned short s) {
  return __uint_as_float(((unsigned)s) << 16);
}

// ---------------- gfx1250 async global->LDS copy (ASYNCcnt-tracked) ------
// ISA 15.18.3 opcode 98: GLOBAL_LOAD_ASYNC_TO_LDS_B128, GV mode.
// VDST = per-lane LDS byte address, VADDR = per-lane 64-bit global address.
__device__ __forceinline__ void async_b128(unsigned lds_off, const void* g) {
  asm volatile("global_load_async_to_lds_b128 %0, %1, off"
               :: "v"(lds_off), "v"(g) : "memory");
}
__device__ __forceinline__ void wait_async0() {
  asm volatile("s_wait_asynccnt 0x0" ::: "memory");
}

// =====================================================================
// Tiled WMMA GEMM:  C[M,N] (fp32) (+)= A[M,K] (bf16) * Bt[N,K] (bf16)
// Both operands K-major -> every 16B chunk of an LDS tile is a contiguous
// global->LDS copy, staged with global_load_async_to_lds_b128 and double
// buffering overlapped with the 8 WMMAs per k-step.
// 128x128x32 macro tile, 256 thr = 8 waves (wave32), wave = 32x64.
// Batched via blockIdx.z with per-operand stride + shift (GQA: z>>1 on B).
// =====================================================================
#define BM    128
#define BN    128
#define BK    32
#define PITCH 40   // LDS row pitch in ushorts (80B: 16B-aligned, bank-spread)

__global__ __launch_bounds__(256)
void gemm_wmma_bf16(const unsigned short* __restrict__ A,   // M x K
                    const unsigned short* __restrict__ B,   // N x K
                    float* __restrict__ C,                  // M x N
                    int K, int lda, int ldb, int ldc,
                    long long aBatch, int aShift,
                    long long bBatch, int bShift,
                    long long cBatch, int cShift,
                    int accum)
{
  __shared__ unsigned short sA[2][BM * PITCH];
  __shared__ unsigned short sB[2][BN * PITCH];

  const int t = threadIdx.x;
  const int z = blockIdx.z;
  A += (long long)(z >> aShift) * aBatch;
  B += (long long)(z >> bShift) * bBatch;
  C += (long long)(z >> cShift) * cBatch;

  const int m0 = blockIdx.y * BM;
  const int n0 = blockIdx.x * BN;

  const int lane = t & 31;
  const int warp = t >> 5;
  const int wm = (warp & 3) * 32;   // wave row offset in tile
  const int wn = (warp >> 2) * 64;  // wave col offset in tile
  const int lrow = lane & 15;
  const int lhi  = lane >> 4;

  FragC acc[2][4];
  if (accum) {
    for (int ms = 0; ms < 2; ++ms)
      for (int ns = 0; ns < 4; ++ns) {
        const int row = m0 + wm + ms * 16 + lhi * 8;
        const int col = n0 + wn + ns * 16 + lrow;
        for (int r = 0; r < 8; ++r)
          acc[ms][ns].f[r] = C[(long long)(row + r) * ldc + col];
      }
  } else {
    for (int ms = 0; ms < 2; ++ms)
      for (int ns = 0; ns < 4; ++ns)
        for (int r = 0; r < 8; ++r) acc[ms][ns].f[r] = 0.0f;
  }

  const int nt = K / BK;

  // per-thread staging coords: 512 16B chunks per tile, 2 per thread/operand
  const int rowIdx = t >> 2;        // 0..63 (+64 for second chunk)
  const int kChunk = (t & 3) * 8;   // ushort offset within BK: 0,8,16,24

  // ---- issue async copies for tile `kt` into buffer `buf` ----
#define ISSUE_TILE(kt_, buf_)                                                  \
  {                                                                            \
    const int k0_ = (kt_) * BK;                                                \
    async_b128((unsigned)(size_t)&sA[buf_][rowIdx * PITCH + kChunk],           \
               A + (long long)(m0 + rowIdx) * lda + k0_ + kChunk);             \
    async_b128((unsigned)(size_t)&sA[buf_][(rowIdx + 64) * PITCH + kChunk],    \
               A + (long long)(m0 + rowIdx + 64) * lda + k0_ + kChunk);        \
    async_b128((unsigned)(size_t)&sB[buf_][rowIdx * PITCH + kChunk],           \
               B + (long long)(n0 + rowIdx) * ldb + k0_ + kChunk);             \
    async_b128((unsigned)(size_t)&sB[buf_][(rowIdx + 64) * PITCH + kChunk],    \
               B + (long long)(n0 + rowIdx + 64) * ldb + k0_ + kChunk);        \
  }

  ISSUE_TILE(0, 0);
  wait_async0();
  __syncthreads();

  for (int kt = 0; kt < nt; ++kt) {
    const int cur = kt & 1;

    if (kt + 1 < nt) ISSUE_TILE(kt + 1, cur ^ 1);  // overlap with WMMAs

    // ---- compute on LDS buffer `cur` ----
    FragAB af[2];
    for (int ms = 0; ms < 2; ++ms) {
      // CDNA5 16-bit A 16x32: lanes0-15 -> K{0..7,16..23}, lanes16-31 -> K{8..15,24..31}
      const uint4* p =
          (const uint4*)&sA[cur][(wm + ms * 16 + lrow) * PITCH + lhi * 8];
      const uint4 lo = p[0];  // 16B: k = lhi*8 .. +7
      const uint4 hi = p[2];  // +32B: k = lhi*8+16 .. +23
      af[ms].u[0] = lo.x; af[ms].u[1] = lo.y; af[ms].u[2] = lo.z; af[ms].u[3] = lo.w;
      af[ms].u[4] = hi.x; af[ms].u[5] = hi.y; af[ms].u[6] = hi.z; af[ms].u[7] = hi.w;
    }
    for (int ns = 0; ns < 4; ++ns) {
      // CDNA5 16-bit B 32x16: lanes0-15 N=lane K{0..7,16..23}, lanes16-31 N=lane-16 K{8..15,24..31}
      FragAB bfrag;
      const uint4* p =
          (const uint4*)&sB[cur][(wn + ns * 16 + lrow) * PITCH + lhi * 8];
      const uint4 q0 = p[0];  // k = lhi*8 .. +7
      const uint4 q1 = p[2];  // k = lhi*8+16 .. +23
      bfrag.u[0] = q0.x; bfrag.u[1] = q0.y; bfrag.u[2] = q0.z; bfrag.u[3] = q0.w;
      bfrag.u[4] = q1.x; bfrag.u[5] = q1.y; bfrag.u[6] = q1.z; bfrag.u[7] = q1.w;
      for (int ms = 0; ms < 2; ++ms)
        acc[ms][ns].v = __builtin_amdgcn_wmma_f32_16x16x32_bf16(
            false, af[ms].v, false, bfrag.v, (short)0, acc[ms][ns].v,
            false, false);
    }

    if (kt + 1 < nt) wait_async0();  // next tile fully in LDS
    __syncthreads();
  }
#undef ISSUE_TILE

  // ---- epilogue: C frag layout (VGPR r: M = hi*8 + r, N = lane&15) ----
  for (int ms = 0; ms < 2; ++ms)
    for (int ns = 0; ns < 4; ++ns) {
      const int row = m0 + wm + ms * 16 + lhi * 8;
      const int col = n0 + wn + ns * 16 + lrow;
      for (int r = 0; r < 8; ++r)
        C[(long long)(row + r) * ldc + col] = acc[ms][ns].f[r];
    }
}

// =====================================================================
// Support kernels
// =====================================================================
__global__ void cvt_kernel(const float* __restrict__ x,
                           unsigned short* __restrict__ y, long long n) {
  long long i = (long long)blockIdx.x * blockDim.x + threadIdx.x;
  const long long stride = (long long)gridDim.x * blockDim.x;
  for (; i < n; i += stride) y[i] = f2bf(x[i]);
}

// transpose + convert: x (K,N) fp32 -> y (N,K) bf16, LDS-tiled, both sides coalesced
__global__ __launch_bounds__(256)
void cvt_t_kernel(const float* __restrict__ x, unsigned short* __restrict__ y,
                  int K, int N) {
  __shared__ unsigned short tile[32][33];
  const int n0 = blockIdx.x * 32, k0 = blockIdx.y * 32;
  const int tx = threadIdx.x & 31, ty = threadIdx.x >> 5;  // ty 0..7
  for (int r = 0; r < 32; r += 8)
    tile[ty + r][tx] = f2bf(x[(long long)(k0 + ty + r) * N + n0 + tx]);
  __syncthreads();
  for (int r = 0; r < 32; r += 8)
    y[(long long)(n0 + ty + r) * K + k0 + tx] = tile[tx][ty + r];
}

__global__ __launch_bounds__(256)
void embed_kernel(const int* __restrict__ ids, const unsigned char* __restrict__ eq,
                  const float* __restrict__ sc, const float* __restrict__ zp,
                  float* __restrict__ h) {
  const int s = blockIdx.x;
  const int id = ids[s];
  const float a = sc[id], b = zp[id];
  const unsigned char* row = eq + (long long)id * DM_;
  float* hr = h + (long long)s * DM_;
  for (int j = threadIdx.x; j < DM_; j += 256)
    hr[j] = (float)row[j] * a + b;
}

__global__ __launch_bounds__(256)
void rmsnorm_kernel(const float* __restrict__ x, const float* __restrict__ w,
                    unsigned short* __restrict__ y) {
  __shared__ float red[256];
  const int t = threadIdx.x;
  const long long row = blockIdx.x;
  const float* xr = x + row * DM_;
  float ss = 0.0f;
  for (int j = t; j < DM_; j += 256) { const float v = xr[j]; ss += v * v; }
  red[t] = ss; __syncthreads();
  for (int o = 128; o > 0; o >>= 1) { if (t < o) red[t] += red[t + o]; __syncthreads(); }
  const float rms = rsqrtf(red[0] / (float)DM_ + EPS_);
  unsigned short* yr = y + row * DM_;
  for (int j = t; j < DM_; j += 256) yr[j] = f2bf(w[j] * xr[j] * rms);
}

// RoPE + per-head RMSNorm for Q -> bf16 (S, H*D)
__global__ __launch_bounds__(64)
void rope_q_kernel(const float* __restrict__ qf, const float* __restrict__ qn,
                   unsigned short* __restrict__ qb) {
  __shared__ float red[64];
  const int d = threadIdx.x;        // 0..63
  const int s = blockIdx.x;
  const int hh = blockIdx.y;
  const float* src = qf + (long long)s * (H_ * D_) + hh * D_;
  const float x1 = src[d], x2 = src[d + 64];
  const float inv = __powf(10000.0f, -(float)d * (1.0f / 64.0f));
  const float ang = (float)(HIST_ + s) * inv;
  float sn, c; __sincosf(ang, &sn, &c);
  const float r1 = x1 * c - x2 * sn;
  const float r2 = x2 * c + x1 * sn;
  red[d] = r1 * r1 + r2 * r2; __syncthreads();
  for (int o = 32; o > 0; o >>= 1) { if (d < o) red[d] += red[d + o]; __syncthreads(); }
  const float rms = rsqrtf(red[0] / (float)D_ + EPS_);
  unsigned short* dst = qb + (long long)s * (H_ * D_) + hh * D_;
  dst[d]      = f2bf(qn[d]      * r1 * rms);
  dst[d + 64] = f2bf(qn[d + 64] * r2 * rms);
}

// RoPE + RMSNorm for K; fp32 save_k (KV,D,T) col HIST+s + bf16 K-major (KV,T,D)
__global__ __launch_bounds__(64)
void rope_k_kernel(const float* __restrict__ kf, const float* __restrict__ kn,
                   float* __restrict__ outK, unsigned short* __restrict__ kall) {
  __shared__ float red[64];
  const int d = threadIdx.x;
  const int s = blockIdx.x;
  const int kv = blockIdx.y;
  const float* src = kf + (long long)s * (KV_ * D_) + kv * D_;
  const float x1 = src[d], x2 = src[d + 64];
  const float inv = __powf(10000.0f, -(float)d * (1.0f / 64.0f));
  const float ang = (float)(HIST_ + s) * inv;
  float sn, c; __sincosf(ang, &sn, &c);
  const float r1 = x1 * c - x2 * sn;
  const float r2 = x2 * c + x1 * sn;
  red[d] = r1 * r1 + r2 * r2; __syncthreads();
  for (int o = 32; o > 0; o >>= 1) { if (d < o) red[d] += red[d + o]; __syncthreads(); }
  const float rms = rsqrtf(red[0] / (float)D_ + EPS_);
  const float o1 = kn[d]      * r1 * rms;
  const float o2 = kn[d + 64] * r2 * rms;
  outK[((long long)kv * D_ + d)      * T_ + (HIST_ + s)] = o1;
  outK[((long long)kv * D_ + d + 64) * T_ + (HIST_ + s)] = o2;
  unsigned short* krow = kall + ((long long)kv * T_ + HIST_ + s) * D_;
  krow[d]      = f2bf(o1);
  krow[d + 64] = f2bf(o2);
}

__global__ __launch_bounds__(256)
void copyk_kernel(const float* __restrict__ kc, float* __restrict__ outK,
                  unsigned short* __restrict__ kall) {
  const int d = blockIdx.x, kv = blockIdx.y;
  const float* src = kc + ((long long)kv * D_ + d) * HIST_;
  const long long dstF = ((long long)kv * D_ + d) * T_;
  for (int j = threadIdx.x; j < HIST_; j += 256) {
    const float v = src[j];
    outK[dstF + j] = v;
    kall[((long long)kv * T_ + j) * D_ + d] = f2bf(v);   // bf16 (T,D) K-major
  }
}

__global__ __launch_bounds__(128)
void copyv_kernel(const float* __restrict__ vc, float* __restrict__ outV,
                  unsigned short* __restrict__ vall) {
  const int row = blockIdx.x, kv = blockIdx.y, d = threadIdx.x;
  const float v = vc[((long long)kv * HIST_ + row) * D_ + d];
  outV[((long long)kv * T_ + row) * D_ + d] = v;
  vall[((long long)kv * D_ + d) * T_ + row] = f2bf(v);   // bf16 (D,T) K-major
}

__global__ __launch_bounds__(128)
void storev_kernel(const float* __restrict__ vf, float* __restrict__ outV,
                   unsigned short* __restrict__ vall) {
  const int s = blockIdx.x, kv = blockIdx.y, d = threadIdx.x;
  const float v = vf[(long long)s * (KV_ * D_) + kv * D_ + d];
  outV[((long long)kv * T_ + HIST_ + s) * D_ + d] = v;
  vall[((long long)kv * D_ + d) * T_ + (HIST_ + s)] = f2bf(v);
}

// causal-bias softmax over T, scores fp32 -> attn bf16
__global__ __launch_bounds__(256)
void softmax_kernel(const float* __restrict__ sc, unsigned short* __restrict__ at) {
  __shared__ float red[256];
  const int t = threadIdx.x, s = blockIdx.x, h = blockIdx.y;
  const float* row = sc + ((long long)h * S_ + s) * T_;
  float m = -1e30f;
  for (int j = t; j < T_; j += 256)
    m = fmaxf(m, row[j] + ((j <= s) ? 0.0f : -128.0f));
  red[t] = m; __syncthreads();
  for (int o = 128; o > 0; o >>= 1) { if (t < o) red[t] = fmaxf(red[t], red[t + o]); __syncthreads(); }
  m = red[0]; __syncthreads();
  float sum = 0.0f;
  for (int j = t; j < T_; j += 256)
    sum += __expf(row[j] + ((j <= s) ? 0.0f : -128.0f) - m);
  red[t] = sum; __syncthreads();
  for (int o = 128; o > 0; o >>= 1) { if (t < o) red[t] += red[t + o]; __syncthreads(); }
  const float inv = 1.0f / red[0];
  unsigned short* orow = at + ((long long)h * S_ + s) * T_;
  for (int j = t; j < T_; j += 256)
    orow[j] = f2bf(__expf(row[j] + ((j <= s) ? 0.0f : -128.0f) - m) * inv);
}

__global__ void silu_kernel(const float* __restrict__ g, const float* __restrict__ u,
                            unsigned short* __restrict__ y, long long n) {
  long long i = (long long)blockIdx.x * blockDim.x + threadIdx.x;
  const long long stride = (long long)gridDim.x * blockDim.x;
  for (; i < n; i += stride) {
    const float gv = g[i];
    y[i] = f2bf((gv / (1.0f + __expf(-gv))) * u[i]);
  }
}

__global__ __launch_bounds__(128)
void gemv_logits(const unsigned short* __restrict__ v,
                 const unsigned short* __restrict__ W, float* __restrict__ out) {
  const int col = blockIdx.x * 128 + threadIdx.x;
  float sum = 0.0f;
  for (int k = 0; k < DM_; ++k)
    sum += bf2f(v[k]) * bf2f(W[(long long)k * V_ + col]);
  out[col] = sum;
}

// =====================================================================
// Host orchestration
// =====================================================================
static inline void launch_gemm(const unsigned short* A, const unsigned short* B,
                               float* C, int M, int N, int K,
                               int lda, int ldb, int ldc,
                               long long aB, int aS, long long bB, int bS,
                               long long cB, int cS, int accum, int Z,
                               hipStream_t stream) {
  dim3 g(N / BN, M / BM, Z);
  gemm_wmma_bf16<<<g, 256, 0, stream>>>(A, B, C, K, lda, ldb, ldc,
                                        aB, aS, bB, bS, cB, cS, accum);
}

static inline void launch_cvt_t(const float* x, unsigned short* y, int K, int N,
                                hipStream_t stream) {
  cvt_t_kernel<<<dim3(N / 32, K / 32), 256, 0, stream>>>(x, y, K, N);
}

extern "C" void kernel_launch(void* const* d_in, const int* in_sizes, int n_in,
                              void* d_out, int out_size, void* d_ws, size_t ws_size,
                              hipStream_t stream) {
  (void)in_sizes; (void)n_in; (void)out_size; (void)ws_size;

  const int*            ids      = (const int*)d_in[0];
  const float*          k_cache  = (const float*)d_in[1];
  const float*          v_cache  = (const float*)d_in[2];
  const unsigned char*  embed_q  = (const unsigned char*)d_in[3];
  const float*          embed_sc = (const float*)d_in[4];
  const float*          embed_zp = (const float*)d_in[5];
  const float*          ln1_w    = (const float*)d_in[6];
  const float*          q_w      = (const float*)d_in[7];
  const float*          k_w      = (const float*)d_in[8];
  const float*          v_w      = (const float*)d_in[9];
  const float*          qn_w     = (const float*)d_in[10];
  const float*          kn_w     = (const float*)d_in[11];
  const float*          o_w      = (const float*)d_in[12];
  const float*          ln2_w    = (const float*)d_in[13];
  const float*          gate_w   = (const float*)d_in[14];
  const float*          up_w     = (const float*)d_in[15];
  const float*          down_w   = (const float*)d_in[16];
  const float*          norm_w   = (const float*)d_in[17];
  const float*          lm_w     = (const float*)d_in[18];

  // ---- workspace carve (256B aligned) ----
  char* wsb = (char*)d_ws;
  size_t off = 0;
  auto take = [&](size_t bytes) -> char* {
    char* p = wsb + off;
    off += (bytes + 255) & ~(size_t)255;
    return p;
  };
  float*          h      = (float*)take((size_t)S_ * DM_ * 4);
  unsigned short* hn     = (unsigned short*)take((size_t)S_ * DM_ * 2);
  float*          qf     = (float*)take((size_t)S_ * H_ * D_ * 4);
  float*          kf     = (float*)take((size_t)S_ * KV_ * D_ * 4);
  float*          vf     = (float*)take((size_t)S_ * KV_ * D_ * 4);
  unsigned short* qb     = (unsigned short*)take((size_t)S_ * H_ * D_ * 2);
  unsigned short* kall   = (unsigned short*)take((size_t)KV_ * T_ * D_ * 2);  // (KV,T,D)
  unsigned short* vall   = (unsigned short*)take((size_t)KV_ * D_ * T_ * 2);  // (KV,D,T)
  float*          scores = (float*)take((size_t)H_ * S_ * T_ * 4);
  unsigned short* attn   = (unsigned short*)take((size_t)H_ * S_ * T_ * 2);
  float*          ctxf   = (float*)take((size_t)S_ * H_ * D_ * 4);
  unsigned short* ctxb   = (unsigned short*)take((size_t)S_ * H_ * D_ * 2);
  float*          gatef  = (float*)take((size_t)S_ * FF_ * 4);
  float*          upf    = (float*)take((size_t)S_ * FF_ * 4);
  unsigned short* actb   = (unsigned short*)take((size_t)S_ * FF_ * 2);
  unsigned short* wq     = (unsigned short*)take((size_t)DM_ * H_ * D_ * 2);  // (H*D, DM)
  unsigned short* wk     = (unsigned short*)take((size_t)DM_ * KV_ * D_ * 2); // (KV*D, DM)
  unsigned short* wv     = (unsigned short*)take((size_t)DM_ * KV_ * D_ * 2);
  unsigned short* wo     = (unsigned short*)take((size_t)H_ * D_ * DM_ * 2);  // (DM, H*D)
  unsigned short* wg     = (unsigned short*)take((size_t)DM_ * FF_ * 2);      // (FF, DM)
  unsigned short* wu     = (unsigned short*)take((size_t)DM_ * FF_ * 2);
  unsigned short* wd     = (unsigned short*)take((size_t)FF_ * DM_ * 2);      // (DM, FF)
  unsigned short* wlm    = (unsigned short*)take((size_t)DM_ * V_ * 2);       // (DM, V)
  unsigned short* lastb  = (unsigned short*)take((size_t)DM_ * 2);

  float* outLogits = (float*)d_out;
  const size_t kvSz = (size_t)KV_ * D_ * T_;  // per-layer save_k/save_v elements
  float* outKbase = outLogits + V_;
  float* outVbase = outKbase + (size_t)L_ * kvSz;

  // ---- embedding dequant ----
  embed_kernel<<<S_, 256, 0, stream>>>(ids, embed_q, embed_sc, embed_zp, h);

  for (int i = 0; i < L_; ++i) {
    float* outK = outKbase + (size_t)i * kvSz;
    float* outV = outVbase + (size_t)i * kvSz;

    // weights -> bf16, transposed to (N,K) so GEMM tiles are K-contiguous
    launch_cvt_t(q_w    + (size_t)i * DM_ * H_ * D_,  wq, DM_, H_ * D_, stream);
    launch_cvt_t(k_w    + (size_t)i * DM_ * KV_ * D_, wk, DM_, KV_ * D_, stream);
    launch_cvt_t(v_w    + (size_t)i * DM_ * KV_ * D_, wv, DM_, KV_ * D_, stream);
    launch_cvt_t(o_w    + (size_t)i * H_ * D_ * DM_,  wo, H_ * D_, DM_, stream);
    launch_cvt_t(gate_w + (size_t)i * DM_ * FF_,      wg, DM_, FF_, stream);
    launch_cvt_t(up_w   + (size_t)i * DM_ * FF_,      wu, DM_, FF_, stream);
    launch_cvt_t(down_w + (size_t)i * FF_ * DM_,      wd, FF_, DM_, stream);

    // hn = rmsnorm(h, ln1) -> bf16
    rmsnorm_kernel<<<S_, 256, 0, stream>>>(h, ln1_w + (size_t)i * DM_, hn);

    // q/k/v projections (WMMA)
    launch_gemm(hn, wq, qf, S_, H_ * D_, DM_, DM_, DM_, H_ * D_, 0, 0, 0, 0, 0, 0, 0, 1, stream);
    launch_gemm(hn, wk, kf, S_, KV_ * D_, DM_, DM_, DM_, KV_ * D_, 0, 0, 0, 0, 0, 0, 0, 1, stream);
    launch_gemm(hn, wv, vf, S_, KV_ * D_, DM_, DM_, DM_, KV_ * D_, 0, 0, 0, 0, 0, 0, 0, 1, stream);

    // RoPE + q/k norm; KV-cache concat into d_out + bf16 attention operands
    rope_q_kernel<<<dim3(S_, H_), 64, 0, stream>>>(qf, qn_w + (size_t)i * D_, qb);
    copyk_kernel<<<dim3(D_, KV_), 256, 0, stream>>>(k_cache + (size_t)i * kvSz / 2, outK, kall);
    rope_k_kernel<<<dim3(S_, KV_), 64, 0, stream>>>(kf, kn_w + (size_t)i * D_, outK, kall);
    copyv_kernel<<<dim3(HIST_, KV_), 128, 0, stream>>>(v_cache + (size_t)i * kvSz / 2, outV, vall);
    storev_kernel<<<dim3(S_, KV_), 128, 0, stream>>>(vf, outV, vall);

    // scores[h] = q_h (S,D) x K_kv^T  — B = kall (T,D) K-major, GQA via z>>1
    launch_gemm(qb, kall, scores, S_, T_, D_,
                H_ * D_, D_, T_,
                /*aB=*/D_, 0, /*bB=*/(long long)T_ * D_, 1,
                /*cB=*/(long long)S_ * T_, 0, 0, H_, stream);

    softmax_kernel<<<dim3(S_, H_), 256, 0, stream>>>(scores, attn);

    // ctx[h] = attn_h (S,T) x V_kv — B = vall (D,T) K-major
    launch_gemm(attn, vall, ctxf, S_, D_, T_,
                T_, T_, H_ * D_,
                /*aB=*/(long long)S_ * T_, 0, /*bB=*/(long long)D_ * T_, 1,
                /*cB=*/D_, 0, 0, H_, stream);

    // o projection with residual accumulate: h += ctx @ o_w
    cvt_kernel<<<2048, 256, 0, stream>>>(ctxf, ctxb, (long long)S_ * H_ * D_);
    launch_gemm(ctxb, wo, h, S_, DM_, H_ * D_, H_ * D_, H_ * D_, DM_, 0, 0, 0, 0, 0, 0, 1, 1, stream);

    // MLP: h += (silu(hn2@gate) * (hn2@up)) @ down
    rmsnorm_kernel<<<S_, 256, 0, stream>>>(h, ln2_w + (size_t)i * DM_, hn);
    launch_gemm(hn, wg, gatef, S_, FF_, DM_, DM_, DM_, FF_, 0, 0, 0, 0, 0, 0, 0, 1, stream);
    launch_gemm(hn, wu, upf,   S_, FF_, DM_, DM_, DM_, FF_, 0, 0, 0, 0, 0, 0, 0, 1, stream);
    silu_kernel<<<2048, 256, 0, stream>>>(gatef, upf, actb, (long long)S_ * FF_);
    launch_gemm(actb, wd, h, S_, DM_, FF_, FF_, FF_, DM_, 0, 0, 0, 0, 0, 0, 1, 1, stream);
  }

  // final norm on last token + logits GEMV (lm_head kept (DM,V) for coalescing)
  cvt_kernel<<<2048, 256, 0, stream>>>(lm_w, wlm, (long long)DM_ * V_);
  rmsnorm_kernel<<<1, 256, 0, stream>>>(h + (size_t)(S_ - 1) * DM_, norm_w, lastb);
  gemv_logits<<<V_ / 128, 128, 0, stream>>>(lastb, wlm, outLogits);
}
